// transKernelModel_2637109920613
// MI455X (gfx1250) — matrix operand order repaired and checked by reading, not compiled, not run on hardware
//
#include <hip/hip_runtime.h>

// ---------------------------------------------------------------------------
// Problem constants (from the reference)
// ---------------------------------------------------------------------------
#define Bsz   2
#define Cch   3
#define Him   48
#define Wim   48
#define Sseq  2304          // H*W
#define Dmod  64
#define NHn   8
#define HDd   8
#define FFd   2048
#define NLl   8
#define Mrow  4608          // B*S, all token-wise GEMMs flatten batch into M
#define KS    21
#define KK    441           // KS*KS
#define KKpad 448           // 28 N-tiles of 16 (weights zero-padded)
#define EPSv  1e-5f

// ---------------------------------------------------------------------------
// Types & helpers
// ---------------------------------------------------------------------------
typedef __attribute__((ext_vector_type(16))) __bf16   v16bf;
typedef __attribute__((ext_vector_type(8)))  __bf16   v8bf;
typedef __attribute__((ext_vector_type(8)))  float    v8f;
typedef __attribute__((ext_vector_type(4)))  unsigned v4u;

static __device__ __forceinline__ __bf16 f2bf(float f) {
  unsigned u = __builtin_bit_cast(unsigned, f);
  unsigned r = (u + 0x7fffu + ((u >> 16) & 1u)) >> 16;   // round-to-nearest-even
  return __builtin_bit_cast(__bf16, (unsigned short)r);
}

static __device__ __forceinline__ v8bf zero8bf() {
  v4u z = {0u, 0u, 0u, 0u};
  return __builtin_bit_cast(v8bf, z);
}

static __device__ __forceinline__ v16bf cat8(v8bf a, v8bf b) {
  return __builtin_shufflevector(a, b, 0, 1, 2, 3, 4, 5, 6, 7,
                                       8, 9, 10, 11, 12, 13, 14, 15);
}

static __device__ __forceinline__ v8f wmma_bf16(v16bf a, v16bf b, v8f c) {
  // D = A(16x32) * B(32x16) + C(16x16), f32 accumulate
  return __builtin_amdgcn_wmma_f32_16x16x32_bf16(
      /*neg_a=*/false, a, /*neg_b=*/false, b,
      /*c_mod=*/(short)0, c, /*reuse_a=*/false, /*reuse_b=*/false);
}

// A-matrix tile: 16x32 bf16 from row-major [M x lda].  ISA layout: lane(0-15)
// row=lane, halves 0..7 -> K 0..7, 8..15 -> K 16..23; lanes 16-31 K offset +8.
// Two aligned 16B vector loads -> global_load_b128 pair.
static __device__ __forceinline__ v16bf load_a_tile(const __bf16* A, int lda,
                                                    int m0, int k0, int lane) {
  const __bf16* p = A + (size_t)(m0 + (lane & 15)) * lda + k0 + ((lane >> 4) << 3);
  v8bf a0 = *(const v8bf*)p;
  v8bf a1 = *(const v8bf*)(p + 16);
  return cat8(a0, a1);
}

// B-matrix tile: 32x16 of W^T, W row-major [N x K] (weights pre-padded so every
// row n0..n0+15 is valid -> fully unconditional 2x16B loads).
static __device__ __forceinline__ v16bf load_b_wT(const __bf16* W, int K,
                                                  int n0, int k0, int lane) {
  const __bf16* p = W + (size_t)(n0 + (lane & 15)) * K + k0 + ((lane >> 4) << 3);
  v8bf b0 = *(const v8bf*)p;
  v8bf b1 = *(const v8bf*)(p + 16);
  return cat8(b0, b1);
}

// ---------------------------------------------------------------------------
// fp32 -> bf16 conversion (plus zero-padded variant for the 441->448 weight)
// ---------------------------------------------------------------------------
__global__ void k_f32_to_bf16(const float* __restrict__ s, __bf16* __restrict__ d, int n) {
  for (int i = blockIdx.x * 256 + threadIdx.x; i < n; i += gridDim.x * 256)
    d[i] = f2bf(s[i]);
}

__global__ void k_f32_to_bf16_pad(const float* __restrict__ s, __bf16* __restrict__ d,
                                  int n_src, int n_dst) {
  for (int i = blockIdx.x * 256 + threadIdx.x; i < n_dst; i += gridDim.x * 256)
    d[i] = (i < n_src) ? f2bf(s[i]) : __builtin_bit_cast(__bf16, (unsigned short)0);
}

// ---------------------------------------------------------------------------
// conv3x3 + BN + ReLU  ->  y[b, s, d] (fp32 master + bf16 mirror)
// Loads are unconditional (clamped coords); tap masked via cndmask on value.
// ---------------------------------------------------------------------------
__global__ void k_conv_bn_relu(const float* __restrict__ x, const float* __restrict__ w,
                               const float* __restrict__ cb, const float* __restrict__ g,
                               const float* __restrict__ bb, const float* __restrict__ mean,
                               const float* __restrict__ var,
                               float* __restrict__ yf, __bf16* __restrict__ ybf) {
  int tid = blockIdx.x * 256 + threadIdx.x;           // (b, s, d) flat
  if (tid >= Mrow * Dmod) return;
  int d = tid & 63;
  int s = (tid >> 6) % Sseq;
  int b = tid / (Sseq * Dmod);
  int i = s / Wim, j = s % Wim;
  float acc = cb[d];
#pragma unroll
  for (int c = 0; c < 3; ++c)
#pragma unroll
    for (int ky = 0; ky < 3; ++ky)
#pragma unroll
      for (int kx = 0; kx < 3; ++kx) {
        int ii = i + ky - 1, jj = j + kx - 1;
        bool ok = ((unsigned)ii < (unsigned)Him) && ((unsigned)jj < (unsigned)Wim);
        int iic = ok ? ii : 0, jjc = ok ? jj : 0;
        float xv = x[((b * Cch + c) * Him + iic) * Wim + jjc];
        float wv = w[((d * Cch + c) * 3 + ky) * 3 + kx];
        acc += ok ? xv * wv : 0.f;
      }
  acc = (acc - mean[d]) * rsqrtf(var[d] + EPSv) * g[d] + bb[d];
  acc = fmaxf(acc, 0.f);
  yf[tid]  = acc;
  ybf[tid] = f2bf(acc);
}

// ---------------------------------------------------------------------------
// Generic WMMA GEMM:  out[M x N] = A[M x K] * W^T + bias, optional ReLU.
// grid = (M/16, Npad/16), block = 32 (one wave per 16x16 tile).
// ---------------------------------------------------------------------------
__global__ void k_gemm_bias_act(const __bf16* __restrict__ A, const __bf16* __restrict__ W,
                                const float* __restrict__ bias,
                                __bf16* __restrict__ outb, float* __restrict__ outf,
                                int N, int K, int relu) {
  const int lane = threadIdx.x & 31;
  const int m0 = blockIdx.x * 16;
  const int n0 = blockIdx.y * 16;
  v8f acc = {0.f, 0.f, 0.f, 0.f, 0.f, 0.f, 0.f, 0.f};
  for (int k0 = 0; k0 < K; k0 += 32) {
    v16bf a = load_a_tile(A, K, m0, k0, lane);
    v16bf b = load_b_wT(W, K, n0, k0, lane);
    acc = wmma_bf16(a, b, acc);
  }
  int col = n0 + (lane & 15);
  if (col < N) {
    float bi = bias[col];
#pragma unroll
    for (int v = 0; v < 8; ++v) {
      int row = m0 + v + ((lane >> 4) << 3);
      float o = acc[v] + bi;
      if (relu) o = fmaxf(o, 0.f);
      if (outb) outb[(size_t)row * N + col] = f2bf(o);
      else      outf[(size_t)row * N + col] = o;
    }
  }
}

// ---------------------------------------------------------------------------
// Fused WMMA GEMM (+bias) + residual + LayerNorm over D=64.
// Block = 128 (4 waves each owning one N-tile of 16) -> full 16x64 row block in
// LDS -> LN stats without another global pass.  Updates fp32 master + bf16 mirror.
// ---------------------------------------------------------------------------
__global__ void k_gemm_res_ln(const __bf16* __restrict__ A, int K,
                              const __bf16* __restrict__ W, const float* __restrict__ bias,
                              const float* __restrict__ gamma, const float* __restrict__ beta,
                              float* __restrict__ yf, __bf16* __restrict__ ybf) {
  __shared__ float tile[16][68];
  const int lane = threadIdx.x & 31;
  const int wv = threadIdx.x >> 5;
  const int m0 = blockIdx.x * 16;
  const int n0 = wv * 16;
  v8f acc = {0.f, 0.f, 0.f, 0.f, 0.f, 0.f, 0.f, 0.f};
  for (int k0 = 0; k0 < K; k0 += 32) {
    v16bf a = load_a_tile(A, K, m0, k0, lane);
    v16bf b = load_b_wT(W, K, n0, k0, lane);
    acc = wmma_bf16(a, b, acc);
  }
  int col = n0 + (lane & 15);
  float bi = bias[col];
#pragma unroll
  for (int v = 0; v < 8; ++v)
    tile[v + ((lane >> 4) << 3)][col] = acc[v] + bi;
  __syncthreads();
  if (threadIdx.x < 16) {
    int r = threadIdx.x;
    int row = m0 + r;
    float mean = 0.f;
    for (int j = 0; j < Dmod; ++j) {
      float z = tile[r][j] + yf[(size_t)row * Dmod + j];   // residual add
      tile[r][j] = z;
      mean += z;
    }
    mean *= (1.f / Dmod);
    float var = 0.f;
    for (int j = 0; j < Dmod; ++j) { float dd = tile[r][j] - mean; var += dd * dd; }
    var *= (1.f / Dmod);
    float inv = rsqrtf(var + EPSv);
    for (int j = 0; j < Dmod; ++j) {
      float o = (tile[r][j] - mean) * inv * gamma[j] + beta[j];
      yf[(size_t)row * Dmod + j]  = o;
      ybf[(size_t)row * Dmod + j] = f2bf(o);
    }
  }
}

// ---------------------------------------------------------------------------
// Flash attention: one wave per (b, h, q-tile of 16).  HD=8 zero-padded into
// the K=32 WMMA contraction.  Online softmax via width-16 shfl reductions on
// the WMMA C layout.  All tile loads are unconditional vector loads + register
// masking (no EXEC-predicated scalar loads).  P tile bounces C->A layout
// through per-wave LDS; same-wave RAW covered by s_wait_dscnt.
// grid = 288, block = 256 (8 waves).  2304 waves cover 16 (b,h) x 144 q-tiles.
// ---------------------------------------------------------------------------
__global__ void k_attn_flash(const __bf16* __restrict__ qkv, __bf16* __restrict__ aout) {
  __shared__ float pbuf[8][16][17];
  const int wv   = threadIdx.x >> 5;
  const int lane = threadIdx.x & 31;
  const int li   = lane & 15;
  const int hi   = lane >> 4;                 // 0 or 1 (lane half)
  const int wid  = blockIdx.x * 8 + wv;       // 0..2303
  const int bh   = wid / 144;                 // 0..15
  const int qt   = wid - bh * 144;            // q-tile 0..143
  const int b = bh >> 3, h = bh & 7;
  const int tb = b * Sseq;
  const int hq = h * HDd;
  const __bf16 bz = __builtin_bit_cast(__bf16, (unsigned short)0);
  const v8bf zv = zero8bf();
  const float scale = 0.35355339059327373f;   // 1/sqrt(HD)
  const unsigned himask = hi ? 0u : 0xFFFFFFFFu;       // keep q/k only in lanes 0-15
  const unsigned short vmask = (li < 8) ? (unsigned short)0xFFFFu : (unsigned short)0;

  // Q tile in A layout (16 queries x 32, only k<8 real): one aligned 16B load
  // for every lane (address always valid), mask upper half-lanes in registers.
  v16bf qa;
  {
    const __bf16* qp = qkv + (size_t)(tb + qt * 16 + li) * 192 + hq;
    v8bf qv = *(const v8bf*)qp;
    v4u qb = __builtin_bit_cast(v4u, qv);
    qb &= himask;
    qa = cat8(__builtin_bit_cast(v8bf, qb), zv);
  }

  float m[8], l[8];
  v8f oacc = {0.f, 0.f, 0.f, 0.f, 0.f, 0.f, 0.f, 0.f};
#pragma unroll
  for (int v = 0; v < 8; ++v) { m[v] = -1e30f; l[v] = 0.f; }

  for (int kt = 0; kt < 144; ++kt) {
    // K^T tile in B layout: contraction = head dim (8 real of 32), n = key
    v16bf kb;
    {
      const __bf16* kp = qkv + (size_t)(tb + kt * 16 + li) * 192 + 64 + hq;
      v8bf kv = *(const v8bf*)kp;
      v4u kbits = __builtin_bit_cast(v4u, kv);
      kbits &= himask;
      kb = cat8(__builtin_bit_cast(v8bf, kbits), zv);
    }
    v8f zc = {0.f, 0.f, 0.f, 0.f, 0.f, 0.f, 0.f, 0.f};
    v8f s = wmma_bf16(qa, kb, zc);            // 16x16 scores

    // online softmax update per row (row stats replicated across 16 lanes)
#pragma unroll
    for (int v = 0; v < 8; ++v) {
      float sv = s[v] * scale;
      float tm = sv;
      tm = fmaxf(tm, __shfl_xor(tm, 8, 16));
      tm = fmaxf(tm, __shfl_xor(tm, 4, 16));
      tm = fmaxf(tm, __shfl_xor(tm, 2, 16));
      tm = fmaxf(tm, __shfl_xor(tm, 1, 16));
      float nm   = fmaxf(m[v], tm);
      float p    = __expf(sv - nm);
      float corr = __expf(m[v] - nm);
      float ps = p;
      ps += __shfl_xor(ps, 8, 16);
      ps += __shfl_xor(ps, 4, 16);
      ps += __shfl_xor(ps, 2, 16);
      ps += __shfl_xor(ps, 1, 16);
      l[v] = l[v] * corr + ps;
      oacc[v] = oacc[v] * corr;
      m[v] = nm;
      pbuf[wv][v + (hi << 3)][li] = p;        // C-layout -> LDS
    }
    asm volatile("s_wait_dscnt 0x0" ::: "memory");   // same-wave LDS RAW

    // P tile in A layout (16 queries x 32 keys; keys 0..15 real).  Conditions
    // are compile-time only -> contiguous 8-float LDS reads, no divergence.
    v16bf pa;
    const float* pp = &pbuf[wv][li][hi << 3];
#pragma unroll
    for (int i = 0; i < 16; ++i)
      pa[i] = (i < 8) ? f2bf(pp[i]) : bz;

    // V tile in B layout: contraction = key (16 real of 32), n = head dim
    // (8 real).  Clamp lane so every load address is valid, mask in registers.
    v16bf vb;
    {
      const __bf16* vbase = qkv + (size_t)(tb + kt * 16 + (hi << 3)) * 192 + 128 + hq + (lane & 7);
#pragma unroll
      for (int i = 0; i < 16; ++i) {
        unsigned short raw = __builtin_bit_cast(unsigned short, vbase[(size_t)(i & 7) * 192]);
        vb[i] = __builtin_bit_cast(__bf16, (unsigned short)(raw & vmask));
      }
    }
    oacc = wmma_bf16(pa, vb, oacc);
  }

  if (li < 8) {
#pragma unroll
    for (int v = 0; v < 8; ++v) {
      int row = v + (hi << 3);
      int tok = tb + qt * 16 + row;
      aout[(size_t)tok * Dmod + hq + li] = f2bf(oacc[v] / l[v]);
    }
  }
}

// ---------------------------------------------------------------------------
// Row softmax over 441 logits, transposed store into kernel_out [B,441,H,W]
// grid = 4608 rows, block = 128
// ---------------------------------------------------------------------------
__global__ void k_softmax_last(const float* __restrict__ logits, float* __restrict__ kout) {
  __shared__ float red[128];
  const int s = blockIdx.x;
  const int b = s / Sseq, pos = s % Sseq;
  const float* row = logits + (size_t)s * KK;
  float mx = -1e30f;
  for (int p = threadIdx.x; p < KK; p += 128) mx = fmaxf(mx, row[p]);
  red[threadIdx.x] = mx; __syncthreads();
  for (int o = 64; o > 0; o >>= 1) {
    if ((int)threadIdx.x < o) red[threadIdx.x] = fmaxf(red[threadIdx.x], red[threadIdx.x + o]);
    __syncthreads();
  }
  mx = red[0]; __syncthreads();
  float sm = 0.f;
  for (int p = threadIdx.x; p < KK; p += 128) sm += __expf(row[p] - mx);
  red[threadIdx.x] = sm; __syncthreads();
  for (int o = 64; o > 0; o >>= 1) {
    if ((int)threadIdx.x < o) red[threadIdx.x] += red[threadIdx.x + o];
    __syncthreads();
  }
  float inv = 1.f / red[0];
  for (int p = threadIdx.x; p < KK; p += 128)
    kout[((size_t)b * KK + p) * Sseq + pos] = __expf(row[p] - mx) * inv;
}

// ---------------------------------------------------------------------------
// Apply predicted 21x21 kernel with reflect padding: out[b,c,y,x]
// ---------------------------------------------------------------------------
__global__ void k_apply_pred(const float* __restrict__ x, const float* __restrict__ kprob,
                             float* __restrict__ out) {
  int tid = blockIdx.x * 256 + threadIdx.x;   // b,c,y,x flat
  if (tid >= Bsz * Cch * Sseq) return;
  int xc = tid % Wim;
  int yr = (tid / Wim) % Him;
  int c  = (tid / Sseq) % Cch;
  int b  = tid / (Cch * Sseq);
  const float* xi = x + (size_t)(b * Cch + c) * Sseq;
  const float* kp = kprob + (size_t)b * KK * Sseq + (size_t)yr * Wim + xc;
  float acc = 0.f;
  for (int ky = 0; ky < KS; ++ky) {
    int ry = yr + ky - KS / 2;
    if (ry < 0) ry = -ry;                     // reflect (no edge repeat)
    if (ry > Him - 1) ry = 2 * (Him - 1) - ry;
    for (int kx = 0; kx < KS; ++kx) {
      int rx = xc + kx - KS / 2;
      if (rx < 0) rx = -rx;
      if (rx > Wim - 1) rx = 2 * (Wim - 1) - rx;
      acc += xi[ry * Wim + rx] * kp[(size_t)(ky * KS + kx) * Sseq];
    }
  }
  out[tid] = acc;
}

// ---------------------------------------------------------------------------
// Host orchestration
// ---------------------------------------------------------------------------
extern "C" void kernel_launch(void* const* d_in, const int* in_sizes, int n_in,
                              void* d_out, int out_size, void* d_ws, size_t ws_size,
                              hipStream_t stream) {
  const float* x        = (const float*)d_in[0];
  const float* conv1_w  = (const float*)d_in[1];
  const float* conv1_b  = (const float*)d_in[2];
  const float* bn_gamma = (const float*)d_in[3];
  const float* bn_beta  = (const float*)d_in[4];
  const float* bn_mean  = (const float*)d_in[5];
  const float* bn_var   = (const float*)d_in[6];
  const float* attn_in_w  = (const float*)d_in[7];
  const float* attn_in_b  = (const float*)d_in[8];
  const float* attn_out_w = (const float*)d_in[9];
  const float* attn_out_b = (const float*)d_in[10];
  const float* lin1_w   = (const float*)d_in[11];
  const float* lin1_b   = (const float*)d_in[12];
  const float* lin2_w   = (const float*)d_in[13];
  const float* lin2_b   = (const float*)d_in[14];
  const float* ln1_g    = (const float*)d_in[15];
  const float* ln1_b    = (const float*)d_in[16];
  const float* ln2_g    = (const float*)d_in[17];
  const float* ln2_b    = (const float*)d_in[18];
  const float* last_w   = (const float*)d_in[19];
  const float* last_b   = (const float*)d_in[20];
  (void)in_sizes; (void)n_in; (void)out_size; (void)ws_size;

  // Workspace carve-out (~35.6 MB total)
  char* ws = (char*)d_ws;
  size_t off = 0;
  auto carve = [&](size_t bytes) -> void* {
    void* p = ws + off;
    off += (bytes + 255) & ~(size_t)255;
    return p;
  };
  float*  yf     = (float*)  carve((size_t)Mrow * Dmod * 4);
  __bf16* ybf    = (__bf16*) carve((size_t)Mrow * Dmod * 2);
  __bf16* qkv    = (__bf16*) carve((size_t)Mrow * 192  * 2);
  __bf16* aout   = (__bf16*) carve((size_t)Mrow * Dmod * 2);
  __bf16* ff1    = (__bf16*) carve((size_t)Mrow * FFd  * 2);
  float*  logits = (float*)  carve((size_t)Mrow * KK   * 4);
  __bf16* w_in   = (__bf16*) carve((size_t)NLl * 192 * Dmod * 2);
  __bf16* w_out  = (__bf16*) carve((size_t)NLl * Dmod * Dmod * 2);
  __bf16* w1     = (__bf16*) carve((size_t)NLl * FFd * Dmod * 2);
  __bf16* w2     = (__bf16*) carve((size_t)NLl * Dmod * FFd * 2);
  __bf16* w_last = (__bf16*) carve((size_t)KKpad * Dmod * 2);   // zero-padded rows

  auto cvt = [&](const float* s, __bf16* d, int n) {
    k_f32_to_bf16<<<(n + 255) / 256, 256, 0, stream>>>(s, d, n);
  };
  cvt(attn_in_w,  w_in,   NLl * 192 * Dmod);
  cvt(attn_out_w, w_out,  NLl * Dmod * Dmod);
  cvt(lin1_w,     w1,     NLl * FFd * Dmod);
  cvt(lin2_w,     w2,     NLl * Dmod * FFd);
  k_f32_to_bf16_pad<<<(KKpad * Dmod + 255) / 256, 256, 0, stream>>>(
      last_w, w_last, KK * Dmod, KKpad * Dmod);

  // Stage 0: conv + bn + relu -> y
  k_conv_bn_relu<<<(Mrow * Dmod) / 256, 256, 0, stream>>>(
      x, conv1_w, conv1_b, bn_gamma, bn_beta, bn_mean, bn_var, yf, ybf);

  // Stage 1..8: transformer layers
  for (int l = 0; l < NLl; ++l) {
    // QKV projection: [4608x64] x [64x192]
    {
      dim3 g(Mrow / 16, 192 / 16);
      k_gemm_bias_act<<<g, 32, 0, stream>>>(ybf, w_in + (size_t)l * 192 * Dmod,
                                            attn_in_b + l * 192, qkv, nullptr,
                                            192, Dmod, 0);
    }
    // Flash attention
    k_attn_flash<<<288, 256, 0, stream>>>(qkv, aout);
    // Out projection + residual + LN1
    k_gemm_res_ln<<<Mrow / 16, 128, 0, stream>>>(
        aout, Dmod, w_out + (size_t)l * Dmod * Dmod, attn_out_b + l * Dmod,
        ln1_g + l * Dmod, ln1_b + l * Dmod, yf, ybf);
    // FF1: [4608x64] x [64x2048] + ReLU
    {
      dim3 g(Mrow / 16, FFd / 16);
      k_gemm_bias_act<<<g, 32, 0, stream>>>(ybf, w1 + (size_t)l * FFd * Dmod,
                                            lin1_b + l * FFd, ff1, nullptr,
                                            FFd, Dmod, 1);
    }
    // FF2 + residual + LN2: [4608x2048] x [2048x64]
    k_gemm_res_ln<<<Mrow / 16, 128, 0, stream>>>(
        ff1, FFd, w2 + (size_t)l * Dmod * FFd, lin2_b + l * Dmod,
        ln2_g + l * Dmod, ln2_b + l * Dmod, yf, ybf);
  }

  // Final projection to 441 logits (weights padded to 448 rows)
  {
    dim3 g(Mrow / 16, KKpad / 16);
    k_gemm_bias_act<<<g, 32, 0, stream>>>(ybf, w_last, last_b, nullptr, logits,
                                          KK, Dmod, 0);
  }

  float* outf = (float*)d_out;
  float* kout = outf + (size_t)Bsz * Cch * Sseq;   // kernel_out region
  k_softmax_last<<<Mrow, 128, 0, stream>>>(logits, kout);
  k_apply_pred<<<(Bsz * Cch * Sseq + 255) / 256, 256, 0, stream>>>(x, kout, outf);
}
